// VQVAE_20392504722010
// MI455X (gfx1250) — matrix operand order repaired and checked by reading, not compiled, not run on hardware
//
#include <hip/hip_runtime.h>
#include <math.h>

// ---------------------------------------------------------------------------
// Types for CDNA5 WMMA (wave32): 16x16x32 bf16 -> f32
// ---------------------------------------------------------------------------
typedef __attribute__((ext_vector_type(16))) __bf16 bf16x16;
typedef __attribute__((ext_vector_type(8)))  float  f32x8;

union Frag { uint4 u[2]; bf16x16 v; };

// ---------------------------------------------------------------------------
// Generic implicit-GEMM conv, NHWC bf16 activations, f32 accumulation.
// Handles 1x1, 3x3 p1 s1, 4x4 p1 s2, and transpose-conv phases (2x2 taps,
// strided output write) via tap tables + output mapping.
// Wave tile: 32 (spatial rows) x 32 (cout) = 2x2 WMMA register tile.
// Block = 8 waves: 4 stacked on M (128 rows) x 2 on N (64 couts).
// Requires: M multiple of 128, Cout multiple of 64 (or ==64), Cin mult of 32.
// ---------------------------------------------------------------------------
struct ConvP {
  int IH, IW, Cin, lgCin;
  int lgOY, lgOX;        // iteration grid is 2^lgOY x 2^lgOX per batch image
  int sy, sx;            // input stride:  iy = oy*sy + dh[tap]
  int Ktot;              // taps * Cin
  int OH, OW, Cout;      // output tensor dims (NHWC)
  int osy, osx, ooy, oox;// output mapping: y = oy*osy + ooy
  int relu;
  short dh[16], dw[16];
};

__global__ __launch_bounds__(256) void conv_wmma(
    const __bf16* __restrict__ in, const __bf16* __restrict__ wre,
    const float* __restrict__ bias, __bf16* __restrict__ out, ConvP p)
{
  const int lane  = threadIdx.x & 31;
  const int wv    = threadIdx.x >> 5;
  const int mquad = wv & 3;            // which 32-row slab in the 128-row block
  const int nhalf = wv >> 2;           // which 32-col slab in the 64-col block
  const int half  = lane >> 4;         // K-half select (WMMA 16-bit layout)
  const int lid   = lane & 15;

  const int mbase = blockIdx.x * 128 + mquad * 32;
  const int nbase = blockIdx.y * 64 + nhalf * 32;
  const int n0 = nbase + lid;
  const int n1 = nbase + 16 + lid;

  // A rows gathered by this lane (two 16-row tiles)
  const int ma0 = mbase + lid;
  const int ma1 = mbase + 16 + lid;
  const int oxm = (1 << p.lgOX) - 1;
  const int oym = (1 << p.lgOY) - 1;
  const int ox0 = ma0 & oxm, oy0 = (ma0 >> p.lgOX) & oym, bb0 = ma0 >> (p.lgOX + p.lgOY);
  const int ox1 = ma1 & oxm, oy1 = (ma1 >> p.lgOX) & oym, bb1 = ma1 >> (p.lgOX + p.lgOY);

  f32x8 acc00 = {0.f,0.f,0.f,0.f,0.f,0.f,0.f,0.f};
  f32x8 acc01 = acc00, acc10 = acc00, acc11 = acc00;

  const __bf16* wrow0 = wre + (size_t)n0 * p.Ktot;
  const __bf16* wrow1 = wre + (size_t)n1 * p.Ktot;

  for (int k = 0; k < p.Ktot; k += 32) {
    const int k0  = k + half * 8;
    const int tap = k0 >> p.lgCin;          // whole 32-K window is one tap
    const int c0  = k0 & (p.Cin - 1);
    const int dhv = p.dh[tap], dwv = p.dw[tap];

    Frag a0, a1, b0, b1;
    {
      const int iy = oy0 * p.sy + dhv, ix = ox0 * p.sx + dwv;
      if ((unsigned)iy < (unsigned)p.IH && (unsigned)ix < (unsigned)p.IW) {
        const uint4* ap = (const uint4*)(in +
            (((size_t)bb0 * p.IH + iy) * p.IW + ix) * p.Cin + c0);
        a0.u[0] = ap[0];   // K = k0    .. k0+7
        a0.u[1] = ap[2];   // K = k0+16 .. k0+23
      } else { a0.u[0] = make_uint4(0u,0u,0u,0u); a0.u[1] = make_uint4(0u,0u,0u,0u); }
    }
    {
      const int iy = oy1 * p.sy + dhv, ix = ox1 * p.sx + dwv;
      if ((unsigned)iy < (unsigned)p.IH && (unsigned)ix < (unsigned)p.IW) {
        const uint4* ap = (const uint4*)(in +
            (((size_t)bb1 * p.IH + iy) * p.IW + ix) * p.Cin + c0);
        a1.u[0] = ap[0];
        a1.u[1] = ap[2];
      } else { a1.u[0] = make_uint4(0u,0u,0u,0u); a1.u[1] = make_uint4(0u,0u,0u,0u); }
    }
    {
      const uint4* bp0 = (const uint4*)(wrow0 + k + half * 8); // K = k+half*8..+7, k+16+half*8..+7
      const uint4* bp1 = (const uint4*)(wrow1 + k + half * 8);
      b0.u[0] = bp0[0]; b0.u[1] = bp0[2];
      b1.u[0] = bp1[0]; b1.u[1] = bp1[2];
    }
    // prefetch next weight chunk into cache (global_prefetch_b8)
    __builtin_prefetch((const void*)(wrow0 + k + 256), 0, 1);
    __builtin_prefetch((const void*)(wrow1 + k + 256), 0, 1);

    acc00 = __builtin_amdgcn_wmma_f32_16x16x32_bf16(false, a0.v, false, b0.v,
                                                    (short)0, acc00, false, false);
    acc01 = __builtin_amdgcn_wmma_f32_16x16x32_bf16(false, a0.v, false, b1.v,
                                                    (short)0, acc01, false, false);
    acc10 = __builtin_amdgcn_wmma_f32_16x16x32_bf16(false, a1.v, false, b0.v,
                                                    (short)0, acc10, false, false);
    acc11 = __builtin_amdgcn_wmma_f32_16x16x32_bf16(false, a1.v, false, b1.v,
                                                    (short)0, acc11, false, false);
  }

  const float bv0 = bias ? bias[n0] : 0.f;
  const float bv1 = bias ? bias[n1] : 0.f;
  for (int mt = 0; mt < 2; ++mt) {
    const f32x8 accA = mt ? acc10 : acc00;   // column tile n0
    const f32x8 accB = mt ? acc11 : acc01;   // column tile n1
    for (int r = 0; r < 8; ++r) {
      const int m   = mbase + mt * 16 + r + half * 8;  // C/D layout: M = r + 8*half
      const int ox2 = m & oxm;
      const int oy2 = (m >> p.lgOX) & oym;
      const int b2  = m >> (p.lgOX + p.lgOY);
      const int yy  = oy2 * p.osy + p.ooy;
      const int xx  = ox2 * p.osx + p.oox;
      __bf16* op = out + (((size_t)b2 * p.OH + yy) * p.OW + xx) * p.Cout;
      float v0 = accA[r] + bv0;
      float v1 = accB[r] + bv1;
      if (p.relu) { v0 = v0 > 0.f ? v0 : 0.f; v1 = v1 > 0.f ? v1 : 0.f; }
      op[n0] = (__bf16)v0;
      op[n1] = (__bf16)v1;
    }
  }
}

// ---------------------------------------------------------------------------
// Weight repack: (O,I,KH,KW) f32 -> Wre[o][t*I + i] bf16 (K-major, t=kh*KW+kw)
// ---------------------------------------------------------------------------
__global__ void repack_std(const float* __restrict__ w, __bf16* __restrict__ o,
                           int O, int I, int KH, int KW)
{
  const int Ktot = I * KH * KW;
  const long total = (long)O * Ktot;
  const long idx = (long)blockIdx.x * blockDim.x + threadIdx.x;
  if (idx >= total) return;
  const int oo = (int)(idx / Ktot);
  const int r  = (int)(idx % Ktot);
  const int t  = r / I;
  const int ci = r % I;
  const int kh = t / KW, kw = t % KW;
  o[idx] = (__bf16)w[(((long)oo * I + ci) * KH + kh) * KW + kw];
}

// Transposed-conv 4x4 s2 p1 -> 4 phase weights of 2x2 taps.
// in: (I=128, O=128, 4, 4) torch layout. out: [ph][cout][ (th*2+tw)*128 + ci ]
__global__ void repack_up(const float* __restrict__ w, __bf16* __restrict__ o)
{
  const int idx = blockIdx.x * blockDim.x + threadIdx.x;
  if (idx >= 4 * 128 * 512) return;
  const int r  = idx & 511;
  const int co = (idx >> 9) & 127;
  const int ph = idx >> 16;            // 512*128 = 65536 per phase
  const int pr = ph >> 1, pc = ph & 1;
  const int t  = r >> 7;               // tap 0..3
  const int ci = r & 127;
  const int th = t >> 1, tw = t & 1;
  const int kh = (pr ^ 1) + 2 * th;
  const int kw = (pc ^ 1) + 2 * tw;
  o[idx] = (__bf16)w[(((long)ci * 128 + co) * 4 + kh) * 4 + kw];
}

// ---------------------------------------------------------------------------
// BatchNorm (training stats over rows, C = 128 fixed)
// ---------------------------------------------------------------------------
__global__ __launch_bounds__(256) void bn_stats(const __bf16* __restrict__ x,
                                                int Mrows, float* __restrict__ st)
{
  __shared__ float s[256];             // [0..127]=sum, [128..255]=sumsq
  s[threadIdx.x] = 0.f;
  __syncthreads();
  const int total = Mrows * 128;
  const int stride = gridDim.x * blockDim.x;
  for (int i = blockIdx.x * blockDim.x + threadIdx.x; i < total; i += stride) {
    const float v = (float)x[i];
    const int c = i & 127;
    atomicAdd(&s[c], v);
    atomicAdd(&s[128 + c], v * v);
  }
  __syncthreads();
  atomicAdd(&st[threadIdx.x], s[threadIdx.x]);
}

__global__ void bn_apply(__bf16* __restrict__ out, const __bf16* __restrict__ x,
                         const __bf16* __restrict__ res, const float* __restrict__ st,
                         int Mrows, int relu)
{
  const int i = blockIdx.x * blockDim.x + threadIdx.x;
  if (i >= Mrows * 128) return;
  const int c = i & 127;
  const float invM = 1.f / (float)Mrows;
  const float mean = st[c] * invM;
  const float var  = st[128 + c] * invM - mean * mean;
  float v = ((float)x[i] - mean) * rsqrtf(var + 1e-5f);
  if (res) v += (float)res[i];
  if (relu) v = fmaxf(v, 0.f);
  out[i] = (__bf16)v;
}

// ---------------------------------------------------------------------------
// Encoder input conv 1x1 (3->128): NCHW f32 -> NHWC bf16
// ---------------------------------------------------------------------------
__global__ void enc_in_k(const float* __restrict__ x, const float* __restrict__ w,
                         const float* __restrict__ b, __bf16* __restrict__ out)
{
  const int i = blockIdx.x * blockDim.x + threadIdx.x;
  if (i >= 4 * 256 * 256 * 128) return;
  const int co = i & 127;
  const int m  = i >> 7;
  const int px = m & 255;
  const int py = (m >> 8) & 255;
  const int bb = m >> 16;
  float s = b[co];
  for (int ci = 0; ci < 3; ++ci)
    s += x[(((bb * 3 + ci) * 256 + py) << 8) + px] * w[co * 3 + ci];
  out[i] = (__bf16)s;
}

// Decoder output conv 1x1 (128->3) + sigmoid: NHWC bf16 -> NCHW f32
__global__ void dec_out_k(const __bf16* __restrict__ h, const float* __restrict__ w,
                          const float* __restrict__ b, float* __restrict__ rec)
{
  const int i = blockIdx.x * blockDim.x + threadIdx.x;
  if (i >= 4 * 3 * 256 * 256) return;
  const int px = i & 255;
  int t = i >> 8;
  const int py = t & 255;
  t >>= 8;
  const int co = t % 3;
  const int bb = t / 3;
  const __bf16* hp = h + (((size_t)(bb * 256 + py) << 8) + px) * 128;
  float s = b[co];
  for (int ci = 0; ci < 128; ++ci) s += (float)hp[ci] * w[co * 128 + ci];
  rec[i] = 1.f / (1.f + expf(-s));
}

// ---------------------------------------------------------------------------
// VQ: norms, bf16 codebook, WMMA distance + argmin
// ---------------------------------------------------------------------------
__global__ void cb_cvt(const float* __restrict__ cb, __bf16* __restrict__ o)
{
  const int i = blockIdx.x * blockDim.x + threadIdx.x;
  if (i < 1024 * 64) o[i] = (__bf16)cb[i];
}
__global__ void cb_norm_k(const float* __restrict__ cb, float* __restrict__ cn)
{
  const int k = blockIdx.x * blockDim.x + threadIdx.x;
  if (k >= 1024) return;
  float s = 0.f;
  for (int c = 0; c < 64; ++c) { const float v = cb[k * 64 + c]; s += v * v; }
  cn[k] = s;
}
__global__ void z_norm_k(const __bf16* __restrict__ z, float* __restrict__ zn)
{
  const int m = blockIdx.x * blockDim.x + threadIdx.x;
  if (m >= 16384) return;
  float s = 0.f;
  for (int c = 0; c < 64; ++c) { const float v = (float)z[m * 64 + c]; s += v * v; }
  zn[m] = s;
}

__global__ __launch_bounds__(256) void vq_argmin(
    const __bf16* __restrict__ z, const __bf16* __restrict__ cb,
    const float* __restrict__ zn, const float* __restrict__ cbn,
    int* __restrict__ idx)
{
  const int lane  = threadIdx.x & 31;
  const int wv    = threadIdx.x >> 5;
  const int mtile = blockIdx.x * 8 + wv;
  const int half  = lane >> 4;
  const int lid   = lane & 15;
  const int ma    = mtile * 16 + lid;

  Frag a0, a1;
  const uint4* zp = (const uint4*)(z + (size_t)ma * 64);
  a0.u[0] = zp[half];     a0.u[1] = zp[half + 2];   // K window 0..31
  a1.u[0] = zp[half + 4]; a1.u[1] = zp[half + 6];   // K window 32..63

  float bestd[8]; int bestn[8]; float zrow[8];
  for (int r = 0; r < 8; ++r) {
    bestd[r] = 3.0e38f; bestn[r] = 0;
    zrow[r] = zn[mtile * 16 + r + half * 8];
  }

  for (int nt = 0; nt < 64; ++nt) {
    const int n = nt * 16 + lid;
    const uint4* cp = (const uint4*)(cb + (size_t)n * 64);
    Frag b0, b1;
    b0.u[0] = cp[half];     b0.u[1] = cp[half + 2];
    b1.u[0] = cp[half + 4]; b1.u[1] = cp[half + 6];
    f32x8 acc = {0.f,0.f,0.f,0.f,0.f,0.f,0.f,0.f};
    acc = __builtin_amdgcn_wmma_f32_16x16x32_bf16(false, a0.v, false, b0.v,
                                                  (short)0, acc, false, false);
    acc = __builtin_amdgcn_wmma_f32_16x16x32_bf16(false, a1.v, false, b1.v,
                                                  (short)0, acc, false, false);
    const float cn = cbn[n];
    for (int r = 0; r < 8; ++r) {
      const float d = zrow[r] - 2.f * acc[r] + cn;
      if (d < bestd[r]) { bestd[r] = d; bestn[r] = n; }
    }
  }
  // reduce over the 16 lanes holding different n for the same m
  for (int off = 1; off < 16; off <<= 1) {
    for (int r = 0; r < 8; ++r) {
      const float od = __shfl_xor(bestd[r], off, 32);
      const int   on = __shfl_xor(bestn[r], off, 32);
      if (od < bestd[r] || (od == bestd[r] && on < bestn[r])) {
        bestd[r] = od; bestn[r] = on;
      }
    }
  }
  if (lid == 0)
    for (int r = 0; r < 8; ++r)
      idx[mtile * 16 + r + half * 8] = bestn[r];
}

__global__ void finalize_idx(const int* __restrict__ idx, float* __restrict__ counts,
                             float* __restrict__ out_idx)
{
  const int i = blockIdx.x * blockDim.x + threadIdx.x;
  if (i >= 16384) return;
  const int v = idx[i];
  atomicAdd(&counts[v], 1.0f);
  out_idx[i] = (float)v;
}

// loss = commit(0) + cb_l(0) - ent = sum p*log(p)
__global__ __launch_bounds__(256) void entropy_k(const float* __restrict__ counts,
                                                 float* __restrict__ loss_out)
{
  __shared__ float sh[256];
  float s = 0.f;
  for (int k = threadIdx.x; k < 1024; k += 256) s += counts[k] + 1e-6f;
  sh[threadIdx.x] = s; __syncthreads();
  for (int o = 128; o > 0; o >>= 1) {
    if (threadIdx.x < o) sh[threadIdx.x] += sh[threadIdx.x + o];
    __syncthreads();
  }
  const float S = sh[0]; __syncthreads();
  float e = 0.f;
  for (int k = threadIdx.x; k < 1024; k += 256) {
    const float p = (counts[k] + 1e-6f) / S;
    e += p * logf(p);
  }
  sh[threadIdx.x] = e; __syncthreads();
  for (int o = 128; o > 0; o >>= 1) {
    if (threadIdx.x < o) sh[threadIdx.x] += sh[threadIdx.x + o];
    __syncthreads();
  }
  if (threadIdx.x == 0) loss_out[0] = sh[0];
}

// ---------------------------------------------------------------------------
// Host side
// ---------------------------------------------------------------------------
static inline dim3 G1(long n) { return dim3((unsigned)((n + 255) / 256)); }

static void conv_launch(hipStream_t st_, const __bf16* in, const __bf16* wre,
                        const float* bias, __bf16* outp,
                        int Bn, int IH, int IW, int Cin, int lgCin,
                        int lgOY, int lgOX, int sy, int sx,
                        int T, const int* dh, const int* dw,
                        int OH, int OW, int Cout,
                        int osy, int osx, int ooy, int oox, int relu)
{
  ConvP p;
  p.IH = IH; p.IW = IW; p.Cin = Cin; p.lgCin = lgCin;
  p.lgOY = lgOY; p.lgOX = lgOX; p.sy = sy; p.sx = sx;
  p.Ktot = T * Cin; p.OH = OH; p.OW = OW; p.Cout = Cout;
  p.osy = osy; p.osx = osx; p.ooy = ooy; p.oox = oox; p.relu = relu;
  for (int t = 0; t < 16; ++t) {
    p.dh[t] = (short)(t < T ? dh[t] : 0);
    p.dw[t] = (short)(t < T ? dw[t] : 0);
  }
  const int M = Bn << (lgOY + lgOX);
  dim3 g(M / 128, Cout / 64 > 0 ? Cout / 64 : 1, 1);
  conv_wmma<<<g, 256, 0, st_>>>(in, wre, bias, outp, p);
}

extern "C" void kernel_launch(void* const* d_in, const int* in_sizes, int n_in,
                              void* d_out, int out_size, void* d_ws, size_t ws_size,
                              hipStream_t stream) {
  (void)in_sizes; (void)n_in; (void)out_size; (void)ws_size;
  const float* x          = (const float*)d_in[0];
  const float* enc_in_w   = (const float*)d_in[1];
  const float* enc_in_b   = (const float*)d_in[2];
  const float* enc_res_w1 = (const float*)d_in[3];
  const float* enc_res_b1 = (const float*)d_in[4];
  const float* enc_res_w2 = (const float*)d_in[5];
  const float* enc_res_b2 = (const float*)d_in[6];
  const float* enc_down_w = (const float*)d_in[7];
  const float* enc_down_b = (const float*)d_in[8];
  const float* enc_out_w  = (const float*)d_in[9];
  const float* enc_out_b  = (const float*)d_in[10];
  const float* dec_in_w   = (const float*)d_in[11];
  const float* dec_in_b   = (const float*)d_in[12];
  const float* dec_res_w1 = (const float*)d_in[13];
  const float* dec_res_b1 = (const float*)d_in[14];
  const float* dec_res_w2 = (const float*)d_in[15];
  const float* dec_res_b2 = (const float*)d_in[16];
  const float* dec_up_w   = (const float*)d_in[17];
  const float* dec_up_b   = (const float*)d_in[18];
  const float* dec_out_w  = (const float*)d_in[19];
  const float* dec_out_b  = (const float*)d_in[20];
  const float* codebook   = (const float*)d_in[21];
  float* out = (float*)d_out;

  // ---- workspace bump allocator ----
  size_t off = 0;
  auto alloc = [&](size_t bytes) -> void* {
    off = (off + 255) & ~(size_t)255;
    void* p = (char*)d_ws + off;
    off += bytes;
    return p;
  };
  const size_t ACT = (size_t)4 * 256 * 256 * 128 * 2;   // 64 MB
  __bf16* A  = (__bf16*)alloc(ACT);
  __bf16* Bf = (__bf16*)alloc(ACT);
  __bf16* Cf = (__bf16*)alloc(ACT);
  __bf16* Z  = (__bf16*)alloc((size_t)4 * 64 * 64 * 64 * 2);
  __bf16* wr_er1 = (__bf16*)alloc((size_t)2 * 128 * 1152 * 2);
  __bf16* wr_er2 = (__bf16*)alloc((size_t)2 * 128 * 1152 * 2);
  __bf16* wr_ed  = (__bf16*)alloc((size_t)2 * 128 * 2048 * 2);
  __bf16* wr_eo  = (__bf16*)alloc((size_t)64 * 128 * 2);
  __bf16* wr_di  = (__bf16*)alloc((size_t)128 * 64 * 2);
  __bf16* wr_dr1 = (__bf16*)alloc((size_t)2 * 128 * 1152 * 2);
  __bf16* wr_dr2 = (__bf16*)alloc((size_t)2 * 128 * 1152 * 2);
  __bf16* wr_du  = (__bf16*)alloc((size_t)2 * 4 * 128 * 512 * 2);
  __bf16* cbb    = (__bf16*)alloc((size_t)1024 * 64 * 2);
  float*  cbn    = (float*)alloc(1024 * 4);
  float*  zn     = (float*)alloc(16384 * 4);
  float*  st     = (float*)alloc(256 * 4);
  float*  counts = (float*)alloc(1024 * 4);
  int*    idxb   = (int*)alloc(16384 * 4);

  // ---- tap tables ----
  int dh9[16], dw9[16], dh16[16], dw16[16], dh1[16], dw1[16];
  for (int t = 0; t < 16; ++t) { dh9[t]=dw9[t]=dh16[t]=dw16[t]=dh1[t]=dw1[t]=0; }
  for (int t = 0; t < 9;  ++t) { dh9[t]  = t / 3 - 1; dw9[t]  = t % 3 - 1; }
  for (int t = 0; t < 16; ++t) { dh16[t] = t / 4 - 1; dw16[t] = t % 4 - 1; }

  // ---- weight repacks (K-major bf16) ----
  repack_std<<<G1(2L*128*1152), 256, 0, stream>>>(enc_res_w1, wr_er1, 256, 128, 3, 3);
  repack_std<<<G1(2L*128*1152), 256, 0, stream>>>(enc_res_w2, wr_er2, 256, 128, 3, 3);
  repack_std<<<G1(2L*128*2048), 256, 0, stream>>>(enc_down_w, wr_ed,  256, 128, 4, 4);
  repack_std<<<G1(64L*128),     256, 0, stream>>>(enc_out_w,  wr_eo,   64, 128, 1, 1);
  repack_std<<<G1(128L*64),     256, 0, stream>>>(dec_in_w,   wr_di,  128,  64, 1, 1);
  repack_std<<<G1(2L*128*1152), 256, 0, stream>>>(dec_res_w1, wr_dr1, 256, 128, 3, 3);
  repack_std<<<G1(2L*128*1152), 256, 0, stream>>>(dec_res_w2, wr_dr2, 256, 128, 3, 3);
  for (int l = 0; l < 2; ++l)
    repack_up<<<G1(4L*128*512), 256, 0, stream>>>(dec_up_w + (size_t)l * 262144,
                                                  wr_du + (size_t)l * 262144);

  // ---- encoder ----
  enc_in_k<<<G1(4L*256*256*128), 256, 0, stream>>>(x, enc_in_w, enc_in_b, A);

  // layer 0 @ 256x256 (act in A)
  conv_launch(stream, A, wr_er1, enc_res_b1, Bf, 4, 256,256,128,7, 8,8, 1,1, 9, dh9,dw9, 256,256,128, 1,1,0,0, 0);
  hipMemsetAsync(st, 0, 1024, stream);
  bn_stats<<<512, 256, 0, stream>>>(Bf, 262144, st);
  bn_apply<<<G1(262144L*128), 256, 0, stream>>>(Bf, Bf, (const __bf16*)nullptr, st, 262144, 1);
  conv_launch(stream, Bf, wr_er2, enc_res_b2, Cf, 4, 256,256,128,7, 8,8, 1,1, 9, dh9,dw9, 256,256,128, 1,1,0,0, 0);
  hipMemsetAsync(st, 0, 1024, stream);
  bn_stats<<<512, 256, 0, stream>>>(Cf, 262144, st);
  bn_apply<<<G1(262144L*128), 256, 0, stream>>>(A, Cf, A, st, 262144, 1);
  conv_launch(stream, A, wr_ed, enc_down_b, Bf, 4, 256,256,128,7, 7,7, 2,2, 16, dh16,dw16, 128,128,128, 1,1,0,0, 1);

  // layer 1 @ 128x128 (act in Bf)
  conv_launch(stream, Bf, wr_er1 + (size_t)128*1152, enc_res_b1+128, Cf, 4, 128,128,128,7, 7,7, 1,1, 9, dh9,dw9, 128,128,128, 1,1,0,0, 0);
  hipMemsetAsync(st, 0, 1024, stream);
  bn_stats<<<512, 256, 0, stream>>>(Cf, 65536, st);
  bn_apply<<<G1(65536L*128), 256, 0, stream>>>(Cf, Cf, (const __bf16*)nullptr, st, 65536, 1);
  conv_launch(stream, Cf, wr_er2 + (size_t)128*1152, enc_res_b2+128, A, 4, 128,128,128,7, 7,7, 1,1, 9, dh9,dw9, 128,128,128, 1,1,0,0, 0);
  hipMemsetAsync(st, 0, 1024, stream);
  bn_stats<<<512, 256, 0, stream>>>(A, 65536, st);
  bn_apply<<<G1(65536L*128), 256, 0, stream>>>(Bf, A, Bf, st, 65536, 1);
  conv_launch(stream, Bf, wr_ed + (size_t)128*2048, enc_down_b+128, Cf, 4, 128,128,128,7, 6,6, 2,2, 16, dh16,dw16, 64,64,128, 1,1,0,0, 1);

  // enc_out 1x1 (128 -> 64): Cf -> Z
  conv_launch(stream, Cf, wr_eo, enc_out_b, Z, 4, 64,64,128,7, 6,6, 1,1, 1, dh1,dw1, 64,64,64, 1,1,0,0, 0);

  // ---- vector quantize (argmin only matters forward; decoder sees z) ----
  cb_cvt<<<G1(1024L*64), 256, 0, stream>>>(codebook, cbb);
  cb_norm_k<<<4, 256, 0, stream>>>(codebook, cbn);
  z_norm_k<<<64, 256, 0, stream>>>(Z, zn);
  vq_argmin<<<128, 256, 0, stream>>>(Z, cbb, zn, cbn, idxb);
  hipMemsetAsync(counts, 0, 4096, stream);
  finalize_idx<<<64, 256, 0, stream>>>(idxb, counts, out + 786433);
  entropy_k<<<1, 256, 0, stream>>>(counts, out + 786432);

  // ---- decoder (input = z; straight-through forward value) ----
  conv_launch(stream, Z, wr_di, dec_in_b, A, 4, 64,64,64,6, 6,6, 1,1, 1, dh1,dw1, 64,64,128, 1,1,0,0, 0);

  // layer 0 @ 64x64 (act in A)
  conv_launch(stream, A, wr_dr1, dec_res_b1, Bf, 4, 64,64,128,7, 6,6, 1,1, 9, dh9,dw9, 64,64,128, 1,1,0,0, 0);
  hipMemsetAsync(st, 0, 1024, stream);
  bn_stats<<<512, 256, 0, stream>>>(Bf, 16384, st);
  bn_apply<<<G1(16384L*128), 256, 0, stream>>>(Bf, Bf, (const __bf16*)nullptr, st, 16384, 1);
  conv_launch(stream, Bf, wr_dr2, dec_res_b2, Cf, 4, 64,64,128,7, 6,6, 1,1, 9, dh9,dw9, 64,64,128, 1,1,0,0, 0);
  hipMemsetAsync(st, 0, 1024, stream);
  bn_stats<<<512, 256, 0, stream>>>(Cf, 16384, st);
  bn_apply<<<G1(16384L*128), 256, 0, stream>>>(A, Cf, A, st, 16384, 1);
  // upsample 0: A(64^2) -> Bf(128^2), 4 sub-pixel phases of 2x2-tap convs
  for (int ph = 0; ph < 4; ++ph) {
    const int pr = ph >> 1, pc = ph & 1;
    int dhu[16], dwu[16];
    for (int t = 0; t < 16; ++t) { dhu[t] = 0; dwu[t] = 0; }
    for (int t = 0; t < 4; ++t) { dhu[t] = pr - (t >> 1); dwu[t] = pc - (t & 1); }
    conv_launch(stream, A, wr_du + (size_t)ph*65536, dec_up_b, Bf,
                4, 64,64,128,7, 6,6, 1,1, 4, dhu,dwu, 128,128,128, 2,2,pr,pc, 1);
  }

  // layer 1 @ 128x128 (act in Bf)
  conv_launch(stream, Bf, wr_dr1 + (size_t)128*1152, dec_res_b1+128, Cf, 4, 128,128,128,7, 7,7, 1,1, 9, dh9,dw9, 128,128,128, 1,1,0,0, 0);
  hipMemsetAsync(st, 0, 1024, stream);
  bn_stats<<<512, 256, 0, stream>>>(Cf, 65536, st);
  bn_apply<<<G1(65536L*128), 256, 0, stream>>>(Cf, Cf, (const __bf16*)nullptr, st, 65536, 1);
  conv_launch(stream, Cf, wr_dr2 + (size_t)128*1152, dec_res_b2+128, A, 4, 128,128,128,7, 7,7, 1,1, 9, dh9,dw9, 128,128,128, 1,1,0,0, 0);
  hipMemsetAsync(st, 0, 1024, stream);
  bn_stats<<<512, 256, 0, stream>>>(A, 65536, st);
  bn_apply<<<G1(65536L*128), 256, 0, stream>>>(Bf, A, Bf, st, 65536, 1);
  // upsample 1: Bf(128^2) -> Cf(256^2)
  for (int ph = 0; ph < 4; ++ph) {
    const int pr = ph >> 1, pc = ph & 1;
    int dhu[16], dwu[16];
    for (int t = 0; t < 16; ++t) { dhu[t] = 0; dwu[t] = 0; }
    for (int t = 0; t < 4; ++t) { dhu[t] = pr - (t >> 1); dwu[t] = pc - (t & 1); }
    conv_launch(stream, Bf, wr_du + (size_t)(4 + ph)*65536, dec_up_b+128, Cf,
                4, 128,128,128,7, 7,7, 1,1, 4, dhu,dwu, 256,256,128, 2,2,pr,pc, 1);
  }

  // dec_out 1x1 (128 -> 3) + sigmoid -> recon (NCHW f32) in d_out
  dec_out_k<<<G1(786432), 256, 0, stream>>>(Cf, dec_out_w, dec_out_b, out);
}
